// DifferentiablePedalChain_32787780338162
// MI455X (gfx1250) — matrix operand order repaired and checked by reading, not compiled, not run on hardware
//
#include <hip/hip_runtime.h>
#include <hip/hip_bf16.h>

// ---------------------------------------------------------------------------
// DifferentiablePedalChain on MI455X (gfx1250, wave32).
//
//   k_zero   : zero x_pad (bf16 padded signal) and IR accumulation buffer
//   k_gain   : tanh distortion + compressor gain computer (elementwise)
//   k_scan   : sequential attack/release one-pole smoother (64 serial rows)
//   k_apply  : apply smoothed gain; d_out <- (1-mix)*x_c, x_pad <- bf16(x_c)
//   k_ncvt   : f32 -> bf16 padded copy of filterbank noise
//   k_bfir   : pre-expand Toeplitz B fragments of the 12 bandpass FIRs
//   k_fir    : WMMA bf16 grouped FIR conv -> env*gain -> atomic mean into IR
//   k_bconv  : pre-expand per-row Toeplitz B fragments of the 32768-tap IR
//   k_conv   : WMMA bf16 long conv (550 GFLOP); B fragments staged into LDS
//              by the Tensor Data Mover (tensor_load_to_lds + s_wait_tensorcnt,
//              double buffered), consumed via ds_load_b128 by all 8 waves.
// ---------------------------------------------------------------------------

typedef __attribute__((ext_vector_type(16))) __bf16 v16bf;
typedef __attribute__((ext_vector_type(8)))  float  v8f;

#define SRF        48000.0f
#define NBANDS     12
#define IR_LEN     32768
#define NTAPS      511
#define NROWS      64          // BS*CHS
#define TLEN       131072
#define EPSF       1e-8f

#define PAD_L      32768
#define XPADLEN    (PAD_L + TLEN + 512)          // halves per padded x row
#define NLEN       (IR_LEN + NTAPS - 1)          // 33278
#define NPADLEN    33344                         // padded noise row (mult of 8)
#define NCH_FIR    17                            // K-chunks of 32 covering 511 taps
#define NCH_CONV   1025                          // K-chunks covering 32768 taps (+edge)
#define BATCH      8                             // B chunks staged per TDM transfer
#define NCHP       1032                          // padded chunk stride (129 batches)
#define CH_HALVES  512                           // halves per chunk (32 lanes * 16)

#if defined(__has_builtin)
#if __has_builtin(__builtin_amdgcn_tensor_load_to_lds)
#define USE_TDM 1
#endif
#endif
#ifndef USE_TDM
#define USE_TDM 0
#endif

static __device__ __forceinline__ unsigned short f2bf(float f) {
  unsigned int u = __float_as_uint(f);
  u += 0x7fffu + ((u >> 16) & 1u);               // round-to-nearest-even
  return (unsigned short)(u >> 16);
}

struct Frag32B { uint4 lo, hi; };

// Load a v16bf fragment from two 16-byte-aligned 8-half runs.
static __device__ __forceinline__ v16bf load_frag(const unsigned short* p0,
                                                  const unsigned short* p1) {
  Frag32B t;
  t.lo = *(const uint4*)p0;
  t.hi = *(const uint4*)p1;
  return __builtin_bit_cast(v16bf, t);
}

#if USE_TDM
typedef __attribute__((ext_vector_type(4))) unsigned int v4u;
typedef __attribute__((ext_vector_type(8))) int          v8i_;
typedef __attribute__((ext_vector_type(4))) int          v4i_;

// 1-D TDM transfer: ndw DWORDs global -> LDS (D# per CDNA5 ISA ch.8).
// clang-23 toolchain: 6-arg form (g0, g1, g2, g3, g4, cpol).
static __device__ __forceinline__ void tdm_stage(unsigned lds_off,
                                                 const void* gsrc,
                                                 unsigned ndw) {
  unsigned long long ga = (unsigned long long)gsrc;
  v4u g0;
  g0.x = 1u;                                        // count=1, user descriptor
  g0.y = lds_off;                                   // lds_addr (bytes)
  g0.z = (unsigned)(ga & 0xffffffffu);              // global_addr[31:0]
  g0.w = (unsigned)((ga >> 32) & 0x01ffffffu) | (2u << 30);  // addr hi | type=2
  v8i_ g1;
  g1[0] = (int)(2u << 16);                          // data_size=4B, no multicast
  g1[1] = (int)(ndw << 16);                         // tensor_dim0[15:0]
  g1[2] = (int)(1u << 16);                          // tensor_dim0 hi | tensor_dim1=1
  g1[3] = (int)(ndw << 16);                         // tile_dim0
  g1[4] = 0;                                        // tile_dim1/2 unused (1-D)
  g1[5] = (int)ndw;                                 // tensor_dim0_stride lo
  g1[6] = 0;
  g1[7] = 0;
  v4i_ z4 = {0, 0, 0, 0};
  v8i_ z8 = {0, 0, 0, 0, 0, 0, 0, 0};
  __builtin_amdgcn_tensor_load_to_lds(g0, g1, z4, z4, z8, 0);
}
#endif

// ---------------------------------------------------------------------------
__global__ void k_zero(uint4* p, long long n) {
  long long i = (long long)blockIdx.x * blockDim.x + threadIdx.x;
  if (i < n) p[i] = make_uint4(0u, 0u, 0u, 0u);
}

// distortion + compressor gain computer
__global__ void k_gain(const float* __restrict__ x,
                       const float* __restrict__ pdrive,
                       const float* __restrict__ pthr,
                       const float* __restrict__ pratio,
                       const float* __restrict__ pknee,
                       float* __restrict__ xd_out,
                       float* __restrict__ gain_out,
                       long long n) {
  long long i = (long long)blockIdx.x * blockDim.x + threadIdx.x;
  if (i >= n) return;
  float drive = pdrive[0], thr = pthr[0], ratio = pratio[0];
  float knee  = fmaxf(pknee[0], 1e-3f);
  float xd  = tanhf(x[i] * exp10f(drive * (1.0f / 20.0f)));
  float xdb = 20.0f * log10f(fabsf(xd) + EPSF);
  float over = xdb - thr;
  float hk = over + 0.5f * knee;
  float gk = xdb + (1.0f / ratio - 1.0f) * hk * hk / (2.0f * knee);
  float g  = (2.0f * over < -knee) ? xdb
           : ((2.0f * over > knee) ? (thr + over / ratio) : gk);
  xd_out[i]   = xd;
  gain_out[i] = g - xdb;
}

// sequential one-pole attack/release smoother; 64 independent serial rows
__global__ void k_scan(float* __restrict__ gain,
                       const float* __restrict__ patk,
                       const float* __restrict__ prel) {
  int r = threadIdx.x;
  if (r >= NROWS) return;
  float aa = __expf(-1000.0f / (SRF * patk[0]));
  float ar = __expf(-1000.0f / (SRF * prel[0]));
  float prev = 0.0f;
  float* g = gain + (long long)r * TLEN;
  for (int t = 0; t < TLEN; ++t) {
    float cur = g[t];
    float a = (cur < prev) ? aa : ar;
    prev = a * prev + (1.0f - a) * cur;
    g[t] = prev;
  }
}

// apply smoothed gain+makeup; d_out <- (1-mix)*x_c ; x_pad <- bf16(x_c)
__global__ void k_apply(const float* __restrict__ gain,
                        const float* __restrict__ pmakeup,
                        const float* __restrict__ pmix,
                        float* __restrict__ out,
                        unsigned short* __restrict__ xpad,
                        long long n) {
  long long i = (long long)blockIdx.x * blockDim.x + threadIdx.x;
  if (i >= n) return;
  float mk = pmakeup[0], mix = pmix[0];
  float xc = out[i] * exp10f((gain[i] + mk) * (1.0f / 20.0f));
  out[i] = (1.0f - mix) * xc;
  long long row = i / TLEN, t = i - row * TLEN;
  xpad[row * (long long)XPADLEN + PAD_L + t] = f2bf(xc);
}

// noise f32 -> padded bf16
__global__ void k_ncvt(const float* __restrict__ noise,
                       unsigned short* __restrict__ npad,
                       long long n) {
  long long i = (long long)blockIdx.x * blockDim.x + threadIdx.x;
  if (i >= n) return;
  long long row = i / NPADLEN, t = i - row * NPADLEN;
  float v = (t < NLEN) ? noise[row * (long long)NLEN + t] : 0.0f;
  npad[i] = f2bf(v);
}

// Pre-expand FIR Toeplitz B fragments: B[k,j] = h[kb + k - j]
__global__ void k_bfir(const float* __restrict__ fir, unsigned short* __restrict__ bf) {
  int tid = blockIdx.x * blockDim.x + threadIdx.x;
  if (tid >= NBANDS * NCH_FIR * 32) return;
  int lane = tid & 31;
  int c    = (tid >> 5) % NCH_FIR;
  int band = tid / (NCH_FIR * 32);
  int j = lane & 15, klo = (lane >> 4) * 16;
  unsigned short* dst = bf + (size_t)tid * 16;
  for (int e = 0; e < 16; ++e) {
    int tap = c * 32 + klo + e - j;
    float v = (tap >= 0 && tap < NTAPS) ? fir[band * NTAPS + tap] : 0.0f;
    dst[e] = f2bf(v);
  }
}

// Grouped FIR conv via WMMA; epilogue: envelope * gain / 12, atomic-add into IR
__global__ void k_fir(const unsigned short* __restrict__ npad,
                      const unsigned short* __restrict__ bfir,
                      const float* __restrict__ pgains,
                      const float* __restrict__ pdecays,
                      float* __restrict__ irbuf) {
  int wave = threadIdx.x >> 5, lane = threadIdx.x & 31;
  int band = blockIdx.y, row = blockIdx.z;
  long long tb = ((long long)blockIdx.x * 8 + wave) * 256;
  const unsigned short* nrow = npad + ((long long)row * NBANDS + band) * NPADLEN;
  int m = lane & 15, b0 = (lane >> 4) * 16;
  long long abase = tb + 16 * m + b0;
  const unsigned short* bf = bfir + ((size_t)(band * NCH_FIR) * 32 + lane) * 16;
  v8f acc = {0, 0, 0, 0, 0, 0, 0, 0};
  for (int c = 0; c < NCH_FIR; ++c) {
    const unsigned short* ap = nrow + abase + c * 32;
    v16bf a = load_frag(ap, ap + 8);
    v16bf b = load_frag(bf, bf + 8);
    bf += CH_HALVES;
    acc = __builtin_amdgcn_wmma_f32_16x16x32_bf16(false, a, false, b,
                                                  (short)0, acc, false, false);
  }
  float gb = pgains[band];
  float d  = pdecays[band] * 10.0f + 1.0f;
  float sc = gb * (1.0f / NBANDS);
  for (int r = 0; r < 8; ++r) {
    int M = r + ((lane >> 4) << 3);
    long long n = tb + 16 * M + (lane & 15);
    float env = __expf(-d * (float)n * (1.0f / (IR_LEN - 1)));
    atomicAdd(&irbuf[(long long)row * IR_LEN + n], acc[r] * env * sc);
  }
}

// Pre-expand per-row IR Toeplitz B fragments: B[s,j] = ir[kb + 31 + j - s]
// Row stride padded to NCHP chunks so fixed-size TDM batches never run off.
__global__ void k_bconv(const float* __restrict__ irf, unsigned short* __restrict__ bc) {
  long long tid = (long long)blockIdx.x * blockDim.x + threadIdx.x;
  if (tid >= (long long)NROWS * NCH_CONV * 32) return;
  int lane = (int)(tid & 31);
  long long w = tid >> 5;
  int c = (int)(w % NCH_CONV);
  int row = (int)(w / NCH_CONV);
  int j = lane & 15, klo = (lane >> 4) * 16;
  int kb = (c - 1) * 32;
  unsigned short* dst = bc + (((size_t)row * NCHP + c) * 32 + lane) * 16;
  const float* ir = irf + (long long)row * IR_LEN;
  for (int e = 0; e < 16; ++e) {
    int idx = kb + 31 + j - (klo + e);
    float v = (idx >= 0 && idx < IR_LEN) ? ir[idx] : 0.0f;
    dst[e] = f2bf(v);
  }
}

// 32768-tap convolution via WMMA (bf16, f32 acc); d_out += mix * wet.
// B fragments (shared by all 8 waves of the block) staged into LDS by the TDM,
// double-buffered; A fragments loaded straight from L2-resident x_pad.
__global__ void k_conv(const unsigned short* __restrict__ xpad,
                       const unsigned short* __restrict__ bconv,
                       const float* __restrict__ pmix,
                       float* __restrict__ out) {
  __shared__ unsigned short sB[2][BATCH * CH_HALVES];   // 2 x 8 KB
  int wave = threadIdx.x >> 5, lane = threadIdx.x & 31;
  int row = blockIdx.y;
  long long tb = ((long long)blockIdx.x * 8 + wave) * 256;
  const unsigned short* xrow = xpad + (long long)row * XPADLEN;
  int m = lane & 15, b0 = (lane >> 4) * 16;
  long long abase0 = PAD_L + tb + 16 * m + b0 + 32;     // chunk c: offset - c*32
  const unsigned short* brow = bconv + (size_t)row * NCHP * CH_HALVES;
  const int nbat = (NCH_CONV + BATCH - 1) / BATCH;      // 129

#if USE_TDM
  if (wave == 0) {
    tdm_stage((unsigned)(unsigned long long)&sB[0][0], brow, BATCH * CH_HALVES / 2);
  }
#endif

  v8f acc = {0, 0, 0, 0, 0, 0, 0, 0};
  for (int bat = 0; bat < nbat; ++bat) {
    int buf = bat & 1;
#if USE_TDM
    if (wave == 0) {
      if (bat + 1 < nbat) {
        tdm_stage((unsigned)(unsigned long long)&sB[buf ^ 1][0],
                  brow + (size_t)(bat + 1) * BATCH * CH_HALVES,
                  BATCH * CH_HALVES / 2);
        __builtin_amdgcn_s_wait_tensorcnt(1);   // current batch landed
      } else {
        __builtin_amdgcn_s_wait_tensorcnt(0);
      }
    }
    __syncthreads();                            // batch visible to all waves
#else
    {                                           // cooperative VGPR staging
      const uint4* src = (const uint4*)(brow + (size_t)bat * BATCH * CH_HALVES);
      uint4* dst = (uint4*)&sB[buf][0];
      for (int i = threadIdx.x; i < BATCH * CH_HALVES / 8; i += 256)
        dst[i] = src[i];
    }
    __syncthreads();
#endif
    for (int cc = 0; cc < BATCH; ++cc) {
      int c = bat * BATCH + cc;
      if (c >= NCH_CONV) break;                 // block-uniform: EXEC stays full
      const unsigned short* ap = xrow + abase0 - (long long)c * 32;
      __builtin_prefetch((const void*)(ap - 32), 0, 1);   // global_prefetch_b8
      v16bf a = load_frag(ap, ap + 8);
      const unsigned short* bp = &sB[buf][(cc * 32 + lane) * 16];
      v16bf b = load_frag(bp, bp + 8);          // ds_load_b128 x2
      acc = __builtin_amdgcn_wmma_f32_16x16x32_bf16(false, a, false, b,
                                                    (short)0, acc, false, false);
    }
    __syncthreads();                            // done reading buf before reuse
  }

  float mix = pmix[0];
  for (int r = 0; r < 8; ++r) {
    int M = r + ((lane >> 4) << 3);
    long long n = tb + 16 * M + (lane & 15);
    float* o = out + (long long)row * TLEN + n;
    *o += mix * acc[r];
  }
}

// ---------------------------------------------------------------------------
extern "C" void kernel_launch(void* const* d_in, const int* in_sizes, int n_in,
                              void* d_out, int out_size, void* d_ws, size_t ws_size,
                              hipStream_t stream) {
  (void)in_sizes; (void)n_in; (void)out_size; (void)ws_size;
  const float* x       = (const float*)d_in[0];
  const float* drive   = (const float*)d_in[1];
  const float* thr     = (const float*)d_in[2];
  const float* ratio   = (const float*)d_in[3];
  const float* atk     = (const float*)d_in[4];
  const float* rel     = (const float*)d_in[5];
  const float* knee    = (const float*)d_in[6];
  const float* makeup  = (const float*)d_in[7];
  const float* mix     = (const float*)d_in[8];
  const float* bgains  = (const float*)d_in[9];
  const float* bdecays = (const float*)d_in[10];
  const float* noise   = (const float*)d_in[11];
  const float* firbank = (const float*)d_in[12];
  float* out = (float*)d_out;

  // workspace layout
  char* ws = (char*)d_ws;
  size_t off = 0;
  float* gain = (float*)(ws + off);                   off += (size_t)NROWS * TLEN * 4;
  unsigned short* xpad = (unsigned short*)(ws + off); off += (size_t)NROWS * XPADLEN * 2;
  unsigned short* npad = (unsigned short*)(ws + off); off += (size_t)NROWS * NBANDS * NPADLEN * 2;
  float* irbuf = (float*)(ws + off);                  off += (size_t)NROWS * IR_LEN * 4;
  unsigned short* bfir = (unsigned short*)(ws + off); off += (size_t)NBANDS * NCH_FIR * 32 * 16 * 2;
  unsigned short* bconv = (unsigned short*)(ws + off); off += (size_t)NROWS * NCHP * CH_HALVES * 2;

  const long long nElems = (long long)NROWS * TLEN;

  {
    long long n16 = ((long long)NROWS * XPADLEN * 2) / 16;
    k_zero<<<dim3((unsigned)((n16 + 255) / 256)), dim3(256), 0, stream>>>((uint4*)xpad, n16);
    long long n16b = ((long long)NROWS * IR_LEN * 4) / 16;
    k_zero<<<dim3((unsigned)((n16b + 255) / 256)), dim3(256), 0, stream>>>((uint4*)irbuf, n16b);
  }

  k_gain<<<dim3((unsigned)((nElems + 255) / 256)), dim3(256), 0, stream>>>(
      x, drive, thr, ratio, knee, out, gain, nElems);

  k_scan<<<dim3(1), dim3(64), 0, stream>>>(gain, atk, rel);

  k_apply<<<dim3((unsigned)((nElems + 255) / 256)), dim3(256), 0, stream>>>(
      gain, makeup, mix, out, xpad, nElems);

  {
    long long nn = (long long)NROWS * NBANDS * NPADLEN;
    k_ncvt<<<dim3((unsigned)((nn + 255) / 256)), dim3(256), 0, stream>>>(noise, npad, nn);
  }

  {
    int nb = NBANDS * NCH_FIR * 32;
    k_bfir<<<dim3((nb + 255) / 256), dim3(256), 0, stream>>>(firbank, bfir);
  }

  k_fir<<<dim3(16, NBANDS, NROWS), dim3(256), 0, stream>>>(
      npad, bfir, bgains, bdecays, irbuf);

  {
    long long nb = (long long)NROWS * NCH_CONV * 32;
    k_bconv<<<dim3((unsigned)((nb + 255) / 256)), dim3(256), 0, stream>>>(irbuf, bconv);
  }

  k_conv<<<dim3(64, NROWS), dim3(256), 0, stream>>>(xpad, bconv, mix, out);
}
